// MultiDDINOLoss_58016418234490
// MI455X (gfx1250) — matrix operand order repaired and checked by reading, not compiled, not run on hardware
//
#include <hip/hip_runtime.h>

#define DIM   8192
#define FEA   384
#define NB    32
#define NP    49
#define T_ROWS 3136   // 2*NB*NP
#define S_ROWS 7232   // NB*(2*49 + 8*16)
#define CHUNK0 1568   // NB*NP

typedef __attribute__((ext_vector_type(16))) _Float16 v16h;
typedef __attribute__((ext_vector_type(8)))  float    v8f;

// ---------------- block reduction helpers (256 threads, 8 waves) -------------
__device__ __forceinline__ float blk_red_sum(float v, float* sm) {
    #pragma unroll
    for (int o = 16; o > 0; o >>= 1) v += __shfl_xor(v, o, 32);
    __syncthreads();
    if ((threadIdx.x & 31) == 0) sm[threadIdx.x >> 5] = v;
    __syncthreads();
    float r = 0.f;
    #pragma unroll
    for (int i = 0; i < 8; i++) r += sm[i];
    return r;
}

__device__ __forceinline__ float blk_red_max(float v, float* sm) {
    #pragma unroll
    for (int o = 16; o > 0; o >>= 1) v = fmaxf(v, __shfl_xor(v, o, 32));
    __syncthreads();
    if ((threadIdx.x & 31) == 0) sm[threadIdx.x >> 5] = v;
    __syncthreads();
    float r = sm[0];
    #pragma unroll
    for (int i = 1; i < 8; i++) r = fmaxf(r, sm[i]);
    return r;
}

// ---------------- K0: L2-normalize feature rows -> f16 -----------------------
__global__ __launch_bounds__(128) void k_norm(const float* __restrict__ in,
                                              _Float16* __restrict__ out) {
    const int r   = blockIdx.x;
    const int tid = threadIdx.x;
    __shared__ float sm[4];
    const float* src = in + (size_t)r * FEA;
    float v0 = src[tid], v1 = src[tid + 128], v2 = src[tid + 256];
    float ss = v0 * v0 + v1 * v1 + v2 * v2;
    #pragma unroll
    for (int o = 16; o > 0; o >>= 1) ss += __shfl_xor(ss, o, 32);
    if ((tid & 31) == 0) sm[tid >> 5] = ss;
    __syncthreads();
    float tot   = sm[0] + sm[1] + sm[2] + sm[3];
    float scale = 1.f / fmaxf(sqrtf(tot), 1e-12f);
    _Float16* dst = out + (size_t)r * FEA;
    dst[tid]       = (_Float16)(v0 * scale);
    dst[tid + 128] = (_Float16)(v1 * scale);
    dst[tid + 256] = (_Float16)(v2 * scale);
}

// ---------------- K1: teacher centered/sharpened softmax ---------------------
__global__ __launch_bounds__(256) void k_teacher_softmax(
        const float* __restrict__ t_in, const float* __restrict__ center,
        const int* __restrict__ epoch_p, float* __restrict__ t_soft) {
    __shared__ float sm[8];
    const int row = blockIdx.x;
    const int tid = threadIdx.x;
    const int epoch = epoch_p[0];
    const float T = (epoch < 30) ? (0.04f + 0.03f * (float)epoch / 29.0f) : 0.07f;
    const float invT = 1.0f / T;
    const float* src = t_in + (size_t)row * DIM;
    float x[32];
    float mx = -3.4e38f;
    #pragma unroll
    for (int j = 0; j < 32; j++) {
        float v = (src[tid + 256 * j] - center[tid + 256 * j]) * invT;
        x[j] = v;
        mx = fmaxf(mx, v);
    }
    mx = blk_red_max(mx, sm);
    float se = 0.f;
    #pragma unroll
    for (int j = 0; j < 32; j++) { x[j] = expf(x[j] - mx); se += x[j]; }
    se = blk_red_sum(se, sm);
    float rinv = 1.0f / se;
    float* dst = t_soft + (size_t)row * DIM;
    #pragma unroll
    for (int j = 0; j < 32; j++) dst[tid + 256 * j] = x[j] * rinv;
}

// ---------------- K2: WMMA similarity + argmax -------------------------------
// one wave per (iq, b, m-tile); sim tile [16 x 64] via v_wmma_f32_16x16x32_f16
__global__ __launch_bounds__(32) void k_sim_argmax(
        const _Float16* __restrict__ sfn, const _Float16* __restrict__ tfn,
        int* __restrict__ idx) {
    const int task = blockIdx.x;         // 0..1023
    const int iq   = task & 1;
    const int rem  = task >> 1;
    const int b    = rem & 31;
    const int mt   = rem >> 5;           // 0..15
    int ck, m0, Sk, offs;
    if (mt < 4)       { ck = 0;          Sk = 49; offs = 0;                      m0 = mt * 16; }
    else if (mt < 8)  { ck = 1;          Sk = 49; offs = CHUNK0;                 m0 = (mt - 4) * 16; }
    else              { ck = 2 + (mt-8); Sk = 16; offs = 3136 + (mt - 8) * 512;  m0 = 0; }
    if (ck == iq) return;                // uniform for the whole wave

    const int lane  = threadIdx.x;
    const int half  = lane >> 4;
    const int nlane = lane & 15;

    const _Float16* arow = sfn + (size_t)(offs + b * Sk) * FEA;
    const _Float16* brow = tfn + (size_t)(iq * CHUNK0 + b * NP) * FEA;

    v8f c[4] = {};
    for (int kk = 0; kk < 12; kk++) {
        const int K0 = kk * 32;
        // A fragment: 16x32 f16, lane = M row (+half K offset)
        v16h a;
        const int mrow  = m0 + nlane;
        const int mclmp = (mrow < Sk) ? mrow : (Sk - 1);
        #pragma unroll
        for (int e = 0; e < 16; e++) {
            const int v   = e >> 1;
            const int kin = ((v < 4) ? 2 * v : 8 + 2 * v) + (e & 1) + half * 8;
            a[e] = arow[mclmp * FEA + K0 + kin];
        }
        #pragma unroll
        for (int nt = 0; nt < 4; nt++) {
            // B fragment: 32x16 f16, lane = N col, K split by lane half
            v16h bb;
            const int p  = nt * 16 + nlane;
            const int pc = (p < NP) ? p : (NP - 1);
            #pragma unroll
            for (int e = 0; e < 16; e++) {
                const int v   = e >> 1;
                const int kin = ((v < 4) ? 2 * v : 8 + 2 * v) + (e & 1) + half * 8;
                bb[e] = brow[pc * FEA + K0 + kin];
            }
            c[nt] = __builtin_amdgcn_wmma_f32_16x16x32_f16(
                        false, a, false, bb, (short)0, c[nt], false, false);
        }
    }
    // argmax over P for each of the 8 M rows this lane-half owns
    #pragma unroll
    for (int r = 0; r < 8; r++) {
        const int m = r + half * 8;
        float best = -3.0e38f; int bestp = 0;
        #pragma unroll
        for (int nt = 0; nt < 4; nt++) {
            const int p  = nt * 16 + nlane;
            const float vv = (p < NP) ? c[nt][r] : -3.0e38f;
            if (vv > best) { best = vv; bestp = p; }   // ascending p -> first max
        }
        #pragma unroll
        for (int o = 1; o < 16; o <<= 1) {
            float ov = __shfl_xor(best, o, 16);
            int   op = __shfl_xor(bestp, o, 16);
            if (ov > best || (ov == best && op < bestp)) { best = ov; bestp = op; }
        }
        const int mrow = m0 + m;
        if (nlane == 0 && mrow < Sk)
            idx[iq * S_ROWS + offs + b * Sk + mrow] = bestp;
    }
}

// ---------------- K3: fused student logsumexp + gathered CE ------------------
__global__ __launch_bounds__(256) void k_loss(
        const float* __restrict__ s_in, const float* __restrict__ t_soft,
        const int* __restrict__ idx, float* __restrict__ loss_arr) {
    __shared__ float sm[8];
    const int srow = blockIdx.x;
    const int tid  = threadIdx.x;
    int ck, Sk, local;
    if (srow < CHUNK0)      { ck = 0; Sk = 49; local = srow; }
    else if (srow < 3136)   { ck = 1; Sk = 49; local = srow - CHUNK0; }
    else { int t2 = srow - 3136; ck = 2 + t2 / 512; Sk = 16; local = t2 % 512; }
    const int b   = local / Sk;
    const float w = 1.0f / (18.0f * 32.0f * (float)Sk);

    const float* row = s_in + (size_t)srow * DIM;
    float x[32];
    float mx = -3.4e38f;
    #pragma unroll
    for (int j = 0; j < 32; j++) {
        float v = row[tid + 256 * j] * 10.0f;   // / STUDENT_TEMP
        x[j] = v;
        mx = fmaxf(mx, v);
    }
    mx = blk_red_max(mx, sm);
    float se = 0.f;
    #pragma unroll
    for (int j = 0; j < 32; j++) se += expf(x[j] - mx);
    se = blk_red_sum(se, sm);
    const float lse = mx + logf(se);

    float acc = 0.f;
    for (int iq = 0; iq < 2; iq++) {
        if (iq == ck) continue;                 // uniform per block
        const int p  = idx[iq * S_ROWS + srow];
        const size_t g = (size_t)(iq * CHUNK0 + b * NP + p) * DIM;
        float dot = 0.f, ts = 0.f;
        #pragma unroll
        for (int j = 0; j < 32; j++) {
            float tv = t_soft[g + tid + 256 * j];
            dot += tv * x[j];
            ts  += tv;
        }
        dot = blk_red_sum(dot, sm);
        ts  = blk_red_sum(ts, sm);
        acc += w * (lse * ts - dot);            // = w * sum(-t * log_softmax)
    }
    if (tid == 0) loss_arr[srow] = acc;
}

// ---------------- K4: fixed-order final reduction ----------------------------
__global__ __launch_bounds__(256) void k_final(const float* __restrict__ loss_arr,
                                               float* __restrict__ out) {
    __shared__ float sm[256];
    float s = 0.f;
    for (int i = threadIdx.x; i < S_ROWS; i += 256) s += loss_arr[i];
    sm[threadIdx.x] = s;
    __syncthreads();
    #pragma unroll
    for (int o = 128; o > 0; o >>= 1) {
        if (threadIdx.x < o) sm[threadIdx.x] += sm[threadIdx.x + o];
        __syncthreads();
    }
    if (threadIdx.x == 0) out[0] = sm[0];
}

// ---------------- launch -----------------------------------------------------
extern "C" void kernel_launch(void* const* d_in, const int* in_sizes, int n_in,
                              void* d_out, int out_size, void* d_ws, size_t ws_size,
                              hipStream_t stream) {
    (void)in_sizes; (void)n_in; (void)out_size; (void)ws_size;
    const float* s_region = (const float*)d_in[0];
    const float* s_fea    = (const float*)d_in[1];
    const float* t_region = (const float*)d_in[2];
    const float* t_fea    = (const float*)d_in[3];
    const float* center   = (const float*)d_in[4];
    const int*   epoch    = (const int*)d_in[5];
    float* out = (float*)d_out;

    char* ws = (char*)d_ws;
    const size_t OFF_TSOFT = 0;                                    // 102,760,448 B
    const size_t OFF_SFN   = OFF_TSOFT + (size_t)T_ROWS * DIM * 4; // 5,554,176 B
    const size_t OFF_TFN   = OFF_SFN   + (size_t)S_ROWS * FEA * 2; // 2,408,448 B
    const size_t OFF_IDX   = OFF_TFN   + (size_t)T_ROWS * FEA * 2; // 57,856 B
    const size_t OFF_LOSS  = OFF_IDX   + (size_t)2 * S_ROWS * 4;   // 28,928 B
    float*    t_soft  = (float*)(ws + OFF_TSOFT);
    _Float16* sfn     = (_Float16*)(ws + OFF_SFN);
    _Float16* tfn     = (_Float16*)(ws + OFF_TFN);
    int*      idx     = (int*)(ws + OFF_IDX);
    float*    lossarr = (float*)(ws + OFF_LOSS);

    k_norm<<<S_ROWS, 128, 0, stream>>>(s_fea, sfn);
    k_norm<<<T_ROWS, 128, 0, stream>>>(t_fea, tfn);
    k_teacher_softmax<<<T_ROWS, 256, 0, stream>>>(t_region, center, epoch, t_soft);
    k_sim_argmax<<<2 * NB * 16, 32, 0, stream>>>(sfn, tfn, idx);
    k_loss<<<S_ROWS, 256, 0, stream>>>(s_region, t_soft, idx, lossarr);
    k_final<<<1, 256, 0, stream>>>(lossarr, out);
}
